// Transformer_61117384622335
// MI455X (gfx1250) — matrix operand order repaired and checked by reading, not compiled, not run on hardware
//
#include <hip/hip_runtime.h>
#include <hip/hip_bf16.h>
#include <math.h>

// ------------------- model constants (fixed by the reference) -------------------
#define MD   512     // model dim D
#define MH   8       // heads
#define MDK  64      // head dim
#define MS   512     // sequence length (S == T)
#define MB   4       // batch
#define MDFF 2048
#define MV   32000
#define ML   6

typedef __attribute__((ext_vector_type(16))) __bf16 v16bf;
typedef __attribute__((ext_vector_type(8)))  float  v8f;

union FragBF { v16bf v; unsigned int u[8]; };

__device__ __forceinline__ unsigned short f32_bf16(float f) {
  unsigned int u = __float_as_uint(f);
  u += 0x7FFFu + ((u >> 16) & 1u);     // round-to-nearest-even
  return (unsigned short)(u >> 16);
}
__device__ __forceinline__ unsigned int pack_bf16(float lo, float hi) {
  return ((unsigned int)f32_bf16(hi) << 16) | (unsigned int)f32_bf16(lo);
}

// epilogue modes
#define EP_NONE      0
#define EP_GELU      1
#define EP_SPLITHEAD 2   // [B*S, D] -> [B, H, S, DK]
#define EP_MERGEHEAD 3   // batched z = b*H+h, write into [B, S, D] at column h*DK

// C = A(fp32,[M,K]) @ B(fp32) (+bias), bf16 WMMA w/ fp32 accum.
// Block tile 128x64x32, 4 waves; each wave owns a 32x64 strip (2x4 16x16 tiles).
// BT=false: B is [K,N] row-major (ldb=N). BT=true: B is [N,K] row-major (ldb=K).
template<int EP, bool BT>
__global__ __launch_bounds__(128)
void gemm_wmma(const float* __restrict__ A, long sA,
               const float* __restrict__ Bw, long sB,
               const float* __restrict__ bias,
               float* __restrict__ C, long sC,
               int M, int N, int K, int ldb, int ldc)
{
  __shared__ unsigned short Ash[128][40];  // [m][k], padded stride (bank spread)
  __shared__ unsigned short Bsh[64][40];   // [n][k]

  const int z = blockIdx.z;
  const float* Az = A  + (size_t)z * (size_t)sA;
  const float* Bz = Bw + (size_t)z * (size_t)sB;
  float* Cz;
  if (EP == EP_MERGEHEAD)
    Cz = C + (size_t)(z / MH) * (size_t)MS * MD + (size_t)(z % MH) * MDK;
  else
    Cz = C + (size_t)z * (size_t)sC;

  const int rowBase = blockIdx.y * 128;
  const int colBase = blockIdx.x * 64;
  const int t    = threadIdx.x;
  const int wave = t >> 5;
  const int lane = t & 31;
  const int half = lane >> 4;
  const int lr   = lane & 15;

  v8f acc[2][4];
  {
    v8f zer = {0.f,0.f,0.f,0.f,0.f,0.f,0.f,0.f};
    #pragma unroll
    for (int mi = 0; mi < 2; ++mi)
      #pragma unroll
      for (int nt = 0; nt < 4; ++nt) acc[mi][nt] = zer;
  }

  for (int k0 = 0; k0 < K; k0 += 32) {
    if (k0 + 32 < K) {  // warm caches for next K-slab (global_prefetch_b8)
      __builtin_prefetch(&Az[(size_t)(rowBase + t) * K + k0 + 32], 0, 1);
    }
    // ---- stage A tile 128x32: float2 global loads, packed bf16x2 DS stores ----
    #pragma unroll
    for (int i = 0; i < 16; ++i) {
      int idx2 = i * 128 + t;                 // 2048 bf16-pairs
      int r = idx2 >> 4, c2 = (idx2 & 15) << 1;
      const float2 av = *reinterpret_cast<const float2*>(
          &Az[(size_t)(rowBase + r) * K + (k0 + c2)]);
      *reinterpret_cast<unsigned int*>(&Ash[r][c2]) = pack_bf16(av.x, av.y);
    }
    // ---- stage B tile as [n][k] (64x32) ----
    if (!BT) {
      #pragma unroll
      for (int i = 0; i < 8; ++i) {
        int idx2 = i * 128 + t;               // 1024 pairs
        int kk2 = (idx2 >> 6) << 1, n = idx2 & 63;
        float b0 = Bz[(size_t)(k0 + kk2)     * ldb + (colBase + n)];
        float b1 = Bz[(size_t)(k0 + kk2 + 1) * ldb + (colBase + n)];
        *reinterpret_cast<unsigned int*>(&Bsh[n][kk2]) = pack_bf16(b0, b1);
      }
    } else {
      #pragma unroll
      for (int i = 0; i < 8; ++i) {
        int idx2 = i * 128 + t;
        int n = idx2 >> 4, kk2 = (idx2 & 15) << 1;
        const float2 bv = *reinterpret_cast<const float2*>(
            &Bz[(size_t)(colBase + n) * ldb + (k0 + kk2)]);
        *reinterpret_cast<unsigned int*>(&Bsh[n][kk2]) = pack_bf16(bv.x, bv.y);
      }
    }
    __syncthreads();

    // A fragment (ISA 7.12.2, 16-bit A 16x32): row = lane&15,
    // VGPR v holds K = (v>=4)*16 + (lane>=16)*8 + (v%4)*2 (+1)
    FragBF fa[2];
    #pragma unroll
    for (int mi = 0; mi < 2; ++mi) {
      const int arow = wave * 32 + mi * 16 + lr;
      #pragma unroll
      for (int vv = 0; vv < 8; ++vv) {
        int kk = ((vv >> 2) << 4) + (half << 3) + ((vv & 3) << 1);
        fa[mi].u[vv] = *reinterpret_cast<const unsigned int*>(&Ash[arow][kk]);
      }
    }
    // B fragment: column per lane, same K mapping as A: K = (v>=4)*16 + (lane>=16)*8 + (v%4)*2 (+1)
    #pragma unroll
    for (int nt = 0; nt < 4; ++nt) {
      FragBF fb;
      const int bcol = nt * 16 + lr;
      #pragma unroll
      for (int vv = 0; vv < 8; ++vv) {
        int kk = ((vv >> 2) << 4) + (half << 3) + ((vv & 3) << 1);
        fb.u[vv] = *reinterpret_cast<const unsigned int*>(&Bsh[bcol][kk]);
      }
      acc[0][nt] = __builtin_amdgcn_wmma_f32_16x16x32_bf16(
          false, fa[0].v, false, fb.v, (short)0, acc[0][nt], false, false);
      acc[1][nt] = __builtin_amdgcn_wmma_f32_16x16x32_bf16(
          false, fa[1].v, false, fb.v, (short)0, acc[1][nt], false, false);
    }
    __syncthreads();
  }

  // epilogue: C/D layout row = (lane>=16)*8 + j, col = lane&15
  #pragma unroll
  for (int mi = 0; mi < 2; ++mi) {
    const int mrow0 = rowBase + wave * 32 + mi * 16 + half * 8;
    #pragma unroll
    for (int nt = 0; nt < 4; ++nt) {
      const int n = colBase + nt * 16 + lr;
      const float bv = bias ? bias[n] : 0.0f;
      #pragma unroll
      for (int j = 0; j < 8; ++j) {
        const int m = mrow0 + j;
        float val = acc[mi][nt][j] + bv;
        if (EP == EP_GELU)
          val = 0.5f * val * (1.0f + erff(val * 0.70710678118654752f));
        if (EP == EP_SPLITHEAD) {
          size_t off = (((size_t)(m >> 9) * MH + (size_t)(n >> 6)) * MS
                        + (size_t)(m & (MS - 1))) * MDK + (size_t)(n & (MDK - 1));
          Cz[off] = val;
        } else {
          Cz[(size_t)m * ldc + n] = val;
        }
      }
    }
  }
}

// row softmax over scores [B*H*S, S]; scale = 1/sqrt(DK); analytic causal mask
__global__ __launch_bounds__(128)
void softmax_rows(float* __restrict__ sc, int causal, float scale) {
  const int row = blockIdx.x;
  const int q = row & (MS - 1);
  float* p = sc + (size_t)row * MS;
  __shared__ float red[128];
  const int t = threadIdx.x;
  float v[4]; float mx = -3.4e38f;
  #pragma unroll
  for (int i = 0; i < 4; ++i) {
    int c = t + i * 128;
    float s = p[c] * scale;
    if (causal && c > q) s = -1e9f;
    v[i] = s; mx = fmaxf(mx, s);
  }
  red[t] = mx; __syncthreads();
  for (int s = 64; s > 0; s >>= 1) { if (t < s) red[t] = fmaxf(red[t], red[t + s]); __syncthreads(); }
  const float m = red[0]; __syncthreads();
  float sum = 0.f;
  #pragma unroll
  for (int i = 0; i < 4; ++i) { v[i] = expf(v[i] - m); sum += v[i]; }
  red[t] = sum; __syncthreads();
  for (int s = 64; s > 0; s >>= 1) { if (t < s) red[t] += red[t + s]; __syncthreads(); }
  const float inv = 1.0f / red[0];
  #pragma unroll
  for (int i = 0; i < 4; ++i) p[t + i * 128] = v[i] * inv;
}

// out = LayerNorm(x (+ res)) * g + b ; one 512-wide row per block (in-place safe)
__global__ __launch_bounds__(128)
void add_layernorm(const float* __restrict__ xin, const float* __restrict__ res,
                   const float* __restrict__ g, const float* __restrict__ b,
                   float* __restrict__ out) {
  const int row = blockIdx.x;
  const float* xr = xin + (size_t)row * MD;
  __shared__ float red[128];
  const int t = threadIdx.x;
  float v[4]; float s = 0.f;
  #pragma unroll
  for (int i = 0; i < 4; ++i) {
    int c = t + i * 128;
    float u = xr[c] + (res ? res[(size_t)row * MD + c] : 0.0f);
    v[i] = u; s += u;
  }
  red[t] = s; __syncthreads();
  for (int k = 64; k > 0; k >>= 1) { if (t < k) red[t] += red[t + k]; __syncthreads(); }
  const float mean = red[0] * (1.0f / MD); __syncthreads();
  float s2 = 0.f;
  #pragma unroll
  for (int i = 0; i < 4; ++i) { float d = v[i] - mean; s2 += d * d; }
  red[t] = s2; __syncthreads();
  for (int k = 64; k > 0; k >>= 1) { if (t < k) red[t] += red[t + k]; __syncthreads(); }
  const float inv = rsqrtf(red[0] * (1.0f / MD) + 1e-5f);
  #pragma unroll
  for (int i = 0; i < 4; ++i) {
    int c = t + i * 128;
    out[(size_t)row * MD + c] = (v[i] - mean) * inv * g[c] + b[c];
  }
}

// out[b,s,c] = emb[tok[b,s], c] * sqrt(D) + sinusoidal PE(s, c)
__global__ __launch_bounds__(256)
void embed_pe(const int* __restrict__ tok, const float* __restrict__ emb,
              float* __restrict__ out) {
  const int idx = blockIdx.x * 256 + threadIdx.x;   // over MB*MS*MD
  const int c  = idx & (MD - 1);
  const int ts = idx >> 9;                          // b*S + s
  const int s  = ts & (MS - 1);
  const int token = tok[ts];
  const float e = emb[(size_t)token * MD + c] * 22.62741699796952f; // sqrt(512)
  const int i2 = c & ~1;
  const float dv  = expf((float)i2 * (-9.210340371976184f / (float)MD)); // -ln(1e4)/D
  const float ang = (float)s * dv;
  const float pe  = (c & 1) ? cosf(ang) : sinf(ang);
  out[idx] = e + pe;
}

// ------------------------------- host orchestration -------------------------------
extern "C" void kernel_launch(void* const* d_in, const int* in_sizes, int n_in,
                              void* d_out, int out_size, void* d_ws, size_t ws_size,
                              hipStream_t stream) {
  (void)in_sizes; (void)n_in; (void)out_size; (void)ws_size;
  auto F = [&](int i) { return (const float*)d_in[i]; };

  const int* src = (const int*)d_in[0];
  const int* tgt = (const int*)d_in[1];
  // d_in[2] = tgt_mask: exactly tril -> applied analytically in softmax_rows
  const float* src_emb = F(3);
  const float* tgt_emb = F(4);

  struct AttnW { const float *Wq,*bq,*Wk,*bk,*Wv,*bv,*Wo,*bo; };
  struct FfnW  { const float *W1,*b1,*W2,*b2; };
  AttnW encA { F(5),F(6),F(7),F(8),F(9),F(10),F(11),F(12) };
  FfnW  encF { F(13),F(14),F(15),F(16) };
  const float *enc_ln1g=F(17), *enc_ln1b=F(18), *enc_ln2g=F(19), *enc_ln2b=F(20);
  const float *enc_lnfg=F(21), *enc_lnfb=F(22);
  AttnW decS { F(23),F(24),F(25),F(26),F(27),F(28),F(29),F(30) };
  AttnW decC { F(31),F(32),F(33),F(34),F(35),F(36),F(37),F(38) };
  FfnW  decF { F(39),F(40),F(41),F(42) };
  const float *dec_ln1g=F(43), *dec_ln1b=F(44), *dec_ln2g=F(45), *dec_ln2b=F(46);
  const float *dec_ln3g=F(47), *dec_ln3b=F(48);
  const float *dec_lnfg=F(49), *dec_lnfb=F(50);
  const float *Wout=F(51), *bout=F(52);

  // workspace carving (fp32)
  float* w = (float*)d_ws;
  size_t o = 0;
  auto alloc = [&](size_t n) { float* r = w + o; o += n; return r; };
  const size_t ND = (size_t)MB * MS * MD;
  float* x       = alloc(ND);                      // encoder stream
  float* y       = alloc(ND);                      // decoder stream
  float* enc_out = alloc(ND);
  float* t0      = alloc(ND);                      // sublayer output
  float* ao      = alloc(ND);                      // merged attn heads
  float* qb      = alloc(ND);
  float* kb      = alloc(ND);
  float* vb      = alloc(ND);
  float* hh      = alloc((size_t)MB * MS * MDFF);  // ffn hidden
  float* sc      = alloc((size_t)MB * MH * MS * MS);

  const int Mrows = MB * MS;  // 2048

  auto gemm_nn = [&](const float* A, const float* B, const float* bias,
                     float* C, int N, int K, int ldbv, int ldcv) {
    gemm_wmma<EP_NONE,false><<<dim3(N/64, Mrows/128, 1), dim3(128), 0, stream>>>(
        A, 0, B, 0, bias, C, 0, Mrows, N, K, ldbv, ldcv);
  };
  auto gemm_split = [&](const float* A, const float* Wm, const float* bias, float* C) {
    gemm_wmma<EP_SPLITHEAD,false><<<dim3(MD/64, Mrows/128, 1), dim3(128), 0, stream>>>(
        A, 0, Wm, 0, bias, C, 0, Mrows, MD, MD, MD, 0);
  };
  auto addln = [&](float* xb, const float* res, const float* g, const float* b, float* ob) {
    add_layernorm<<<dim3(MB * MS), dim3(128), 0, stream>>>(xb, res, g, b, ob);
  };
  auto mha = [&](const AttnW& a, int l, const float* qin, const float* kvin,
                 bool causal, float* outbuf) {
    const size_t wo = (size_t)l * MD * MD, bo = (size_t)l * MD;
    gemm_split(qin,  a.Wq + wo, a.bq + bo, qb);
    gemm_split(kvin, a.Wk + wo, a.bk + bo, kb);
    gemm_split(kvin, a.Wv + wo, a.bv + bo, vb);
    // scores[bh] = Q[bh] @ K[bh]^T  (NT, batched over B*H)
    gemm_wmma<EP_NONE,true><<<dim3(MS/64, MS/128, MB*MH), dim3(128), 0, stream>>>(
        qb, (long)MS*MDK, kb, (long)MS*MDK, nullptr, sc, (long)MS*MS,
        MS, MS, MDK, MDK, MS);
    softmax_rows<<<dim3(MB*MH*MS), dim3(128), 0, stream>>>(sc, causal ? 1 : 0, 0.125f);
    // out[bh] = P[bh] @ V[bh], scattered back to [B,S,D]
    gemm_wmma<EP_MERGEHEAD,false><<<dim3(MDK/64, MS/128, MB*MH), dim3(128), 0, stream>>>(
        sc, (long)MS*MS, vb, (long)MS*MDK, nullptr, ao, 0,
        MS, MDK, MS, MDK, MD);
    gemm_nn(ao, a.Wo + wo, a.bo + bo, outbuf, MD, MD, MD, MD);
  };
  auto ffn = [&](const FfnW& f, int l, const float* xin, float* outbuf) {
    gemm_wmma<EP_GELU,false><<<dim3(MDFF/64, Mrows/128, 1), dim3(128), 0, stream>>>(
        xin, 0, f.W1 + (size_t)l*MD*MDFF, 0, f.b1 + (size_t)l*MDFF, hh, 0,
        Mrows, MDFF, MD, MDFF, MDFF);
    gemm_nn(hh, f.W2 + (size_t)l*MDFF*MD, f.b2 + (size_t)l*MD, outbuf, MD, MDFF, MD, MD);
  };

  // ---------------- encoder ----------------
  embed_pe<<<dim3((Mrows * MD) / 256), dim3(256), 0, stream>>>(src, src_emb, x);
  for (int l = 0; l < ML; ++l) {
    mha(encA, l, x, x, false, t0);
    addln(x, t0, enc_ln1g + (size_t)l*MD, enc_ln1b + (size_t)l*MD, x);
    ffn(encF, l, x, t0);
    addln(x, t0, enc_ln2g + (size_t)l*MD, enc_ln2b + (size_t)l*MD, x);
  }
  addln(x, nullptr, enc_lnfg, enc_lnfb, enc_out);

  // ---------------- decoder ----------------
  embed_pe<<<dim3((Mrows * MD) / 256), dim3(256), 0, stream>>>(tgt, tgt_emb, y);
  for (int l = 0; l < ML; ++l) {
    mha(decS, l, y, y, true, t0);
    addln(y, t0, dec_ln1g + (size_t)l*MD, dec_ln1b + (size_t)l*MD, y);
    mha(decC, l, y, enc_out, false, t0);
    addln(y, t0, dec_ln2g + (size_t)l*MD, dec_ln2b + (size_t)l*MD, y);
    ffn(decF, l, y, t0);
    addln(y, t0, dec_ln3g + (size_t)l*MD, dec_ln3b + (size_t)l*MD, y);
  }
  addln(y, nullptr, dec_lnfg, dec_lnfb, y);

  // ---------------- logits: [2048,512] @ [512,32000] + bout ----------------
  gemm_wmma<EP_NONE,false><<<dim3(MV/64, Mrows/128, 1), dim3(128), 0, stream>>>(
      y, 0, Wout, 0, bout, (float*)d_out, 0, Mrows, MV, MD, MV, MV);
}
